// PixNerDiT_44066364457390
// MI455X (gfx1250) — compile-verified, matrix operand order checked by reading, not run on hardware
//
#include <hip/hip_runtime.h>
#include <hip/hip_bf16.h>
#include <cstdint>

// ---------------- problem constants ----------------
#define B_    4
#define N_    2048
#define M_    100
#define C_    1152
#define H_    12
#define HD_   96
#define HALF_ 48
#define FF_   4608
#define NK_   (N_ + M_)          // 2148
#define BN_   (B_ * N_)          // 8192
#define SIXC_ (6 * C_)           // 6912

typedef __bf16 bf16_t;
typedef __attribute__((ext_vector_type(16))) __bf16 v16bf;
typedef __attribute__((ext_vector_type(8)))  __bf16 bfx8;
typedef __attribute__((ext_vector_type(8)))  float  v8f;

// ---------------- CDNA5 async global->LDS (ASYNCcnt-tracked) ----------------
// per-lane copy of 16 bytes from global memory to LDS
__device__ __forceinline__ void async_g2l_b128(uint32_t lds_off, const void* g) {
  asm volatile("global_load_async_to_lds_b128 %0, %1, off"
               :: "v"(lds_off), "v"((unsigned long long)(size_t)g)
               : "memory");
}
__device__ __forceinline__ void wait_asynccnt0() {
  asm volatile("s_wait_asynccnt 0x0" ::: "memory");
}
__device__ __forceinline__ uint32_t lds_off32(const void* p) {
  // generic LDS pointer: low 32 bits are the wave-relative LDS byte address
  return (uint32_t)(size_t)p;
}

// ---------------- WMMA fragment helpers (wave32, 16x16x32 bf16) ----------------
// A-matrix 16x32 (MxK), 16-bit layout per CDNA5 ISA 7.12.2.
__device__ __forceinline__ v16bf frag_a_16x32(const bf16_t* p, int stride) {
  int lane = threadIdx.x & 31;
  int m = lane & 15, hi = lane >> 4;
  v16bf f;
#pragma unroll
  for (int i = 0; i < 8; ++i) {
    int k = (i < 4 ? 2 * i : 16 + 2 * (i - 4)) + hi * 8;
    f[2 * i]     = p[m * stride + k];
    f[2 * i + 1] = p[m * stride + k + 1];
  }
  return f;
}

// B-matrix 32x16 (KxN) stored row-major KxN in LDS: element B[k][n] = p[k*stride + n]
__device__ __forceinline__ v16bf frag_b_32x16(const bf16_t* p, int stride) {
  int lane = threadIdx.x & 31;
  int n = lane & 15, hi = lane >> 4;
  v16bf f;
#pragma unroll
  for (int i = 0; i < 8; ++i) {
    int k = (i < 4 ? 2 * i : 16 + 2 * (i - 4)) + hi * 8;
    f[2 * i]     = p[k * stride + n];
    f[2 * i + 1] = p[(k + 1) * stride + n];
  }
  return f;
}

// B-matrix 32x16 (KxN) where memory holds B^T row-major (N x K): B[k][n] = p[n*stride + k]
__device__ __forceinline__ v16bf frag_b_from_nt(const bf16_t* p, int stride) {
  int lane = threadIdx.x & 31;
  int n = lane & 15, hi = lane >> 4;
  v16bf f;
#pragma unroll
  for (int i = 0; i < 8; ++i) {
    int k = (i < 4 ? 2 * i : 16 + 2 * (i - 4)) + hi * 8;
    f[2 * i]     = p[n * stride + k];
    f[2 * i + 1] = p[n * stride + k + 1];
  }
  return f;
}

__device__ __forceinline__ v8f wmma_bf16(v16bf a, v16bf b, v8f c) {
  return __builtin_amdgcn_wmma_f32_16x16x32_bf16(false, a, false, b, (short)0, c, false, false);
}

// ---------------- small kernels ----------------
__global__ void cast_bf16_kernel(const float* __restrict__ src, bf16_t* __restrict__ dst, int n) {
  int i = blockIdx.x * 256 + threadIdx.x;
  if (i < n) dst[i] = (bf16_t)src[i];
}

__global__ __launch_bounds__(256, 1) void adaln_kernel(const float* __restrict__ cvec,
                                                       const float* __restrict__ w,
                                                       const float* __restrict__ bias,
                                                       float* __restrict__ mods) {
  int t = blockIdx.x * 256 + threadIdx.x;
  if (t >= B_ * SIXC_) return;
  int b = t / SIXC_, j = t % SIXC_;
  const float* cb = cvec + (size_t)b * C_;
  float acc = bias[j];
  for (int k = 0; k < C_; ++k) acc += cb[k] * w[(size_t)k * SIXC_ + j];
  mods[t] = acc;
}

__global__ __launch_bounds__(256, 1) void modulate_kernel(const float* __restrict__ x,
                                                          const float* __restrict__ normw,
                                                          const float* __restrict__ mods,
                                                          int sh_off, int sc_off,
                                                          bf16_t* __restrict__ out) {
  int row = blockIdx.x;
  int tid = threadIdx.x;
  const float* xr = x + (size_t)row * C_;
  float ss = 0.0f;
  for (int c = tid; c < C_; c += 256) { float v = xr[c]; ss += v * v; }
  __shared__ float red[256];
  red[tid] = ss;
  __syncthreads();
  for (int s = 128; s > 0; s >>= 1) {
    if (tid < s) red[tid] += red[tid + s];
    __syncthreads();
  }
  float rsq = rsqrtf(red[0] / (float)C_ + 1e-6f);
  int b = row / N_;
  const float* mb = mods + (size_t)b * SIXC_;
  for (int c = tid; c < C_; c += 256) {
    float hv = xr[c] * rsq * normw[c] * (1.0f + mb[sc_off + c]) + mb[sh_off + c];
    out[(size_t)row * C_ + c] = (bf16_t)hv;
  }
}

__global__ __launch_bounds__(96, 1) void qkv_post_kernel(const float* __restrict__ qkv,
                                                         const float* __restrict__ qnw,
                                                         const float* __restrict__ knw,
                                                         const float* __restrict__ rc,
                                                         const float* __restrict__ rs,
                                                         bf16_t* __restrict__ qb,
                                                         bf16_t* __restrict__ kb,
                                                         bf16_t* __restrict__ vb) {
  int d = threadIdx.x;
  int idx = blockIdx.x;
  int h = idx % H_;
  int bn = idx / H_;
  int n = bn % N_;
  int b = bn / N_;
  const float* base = qkv + (size_t)bn * (3 * C_);
  float qv = base[h * HD_ + d];
  float kv = base[C_ + h * HD_ + d];
  float vv = base[2 * C_ + h * HD_ + d];
  __shared__ float red[2 * HD_];
  __shared__ float sums[2];
  red[d] = qv * qv;
  red[HD_ + d] = kv * kv;
  __syncthreads();
  if (d < 32) {
    float s1 = red[d] + red[d + 32] + red[d + 64];
    float s2 = red[HD_ + d] + red[HD_ + d + 32] + red[HD_ + d + 64];
    for (int off = 16; off > 0; off >>= 1) { s1 += __shfl_xor(s1, off); s2 += __shfl_xor(s2, off); }
    if (d == 0) { sums[0] = s1; sums[1] = s2; }
  }
  __syncthreads();
  float qn = qv * rsqrtf(sums[0] / (float)HD_ + 1e-6f) * qnw[d];
  float kn = kv * rsqrtf(sums[1] / (float)HD_ + 1e-6f) * knw[d];
  red[d] = qn;
  red[HD_ + d] = kn;
  __syncthreads();
  int i = d >> 1;
  float cs = rc[n * HALF_ + i], sn = rs[n * HALF_ + i];
  float qo, ko;
  if ((d & 1) == 0) {
    qo = qn * cs - red[d + 1] * sn;
    ko = kn * cs - red[HD_ + d + 1] * sn;
  } else {
    qo = red[d - 1] * sn + qn * cs;
    ko = red[HD_ + d - 1] * sn + kn * cs;
  }
  size_t bh = (size_t)b * H_ + h;
  qb[(bh * N_ + n) * HD_ + d]  = (bf16_t)qo;
  kb[(bh * NK_ + n) * HD_ + d] = (bf16_t)ko;
  vb[(bh * NK_ + n) * HD_ + d] = (bf16_t)vv;
}

__global__ __launch_bounds__(96, 1) void ykv_post_kernel(const float* __restrict__ ykv,
                                                         const float* __restrict__ knw,
                                                         bf16_t* __restrict__ kb,
                                                         bf16_t* __restrict__ vb) {
  int d = threadIdx.x;
  int idx = blockIdx.x;
  int h = idx % H_;
  int bm = idx / H_;
  int m = bm % M_;
  int b = bm / M_;
  const float* base = ykv + (size_t)bm * (2 * C_);
  float kv = base[h * HD_ + d];
  float vv = base[C_ + h * HD_ + d];
  __shared__ float red[HD_];
  __shared__ float sums[1];
  red[d] = kv * kv;
  __syncthreads();
  if (d < 32) {
    float s1 = red[d] + red[d + 32] + red[d + 64];
    for (int off = 16; off > 0; off >>= 1) s1 += __shfl_xor(s1, off);
    if (d == 0) sums[0] = s1;
  }
  __syncthreads();
  float kn = kv * rsqrtf(sums[0] / (float)HD_ + 1e-6f) * knw[d];
  size_t bh = (size_t)b * H_ + h;
  kb[(bh * NK_ + N_ + m) * HD_ + d] = (bf16_t)kn;
  vb[(bh * NK_ + N_ + m) * HD_ + d] = (bf16_t)vv;
}

// ---------------- tiled bf16 WMMA GEMM (async double-buffered LDS) ----------------
// MODE 0: outf = A@B
// MODE 1: outf = resid + g_msa * (A@B + bias)          (proj epilogue)
// MODE 2: outb = bf16( silu(A@B1) * (A@B2) )           (fused MLP gate)
// MODE 3: outf = resid + g_mlp * (A@B)                 (final epilogue)
template <int MODE>
__global__ __launch_bounds__(256, 1) void gemm_bf16_kernel(
    const bf16_t* __restrict__ A, const bf16_t* __restrict__ B1, const bf16_t* __restrict__ B2p,
    int Mrows, int Ncols, int K,
    float* __restrict__ outf, bf16_t* __restrict__ outb,
    const float* __restrict__ resid, const float* __restrict__ bias,
    const float* __restrict__ mods) {
  constexpr int NB = (MODE == 2) ? 2 : 1;
  __shared__ __align__(16) bf16_t As[2][128][40];
  __shared__ __align__(16) bf16_t Bs[2][NB][32][136];
  int tid = threadIdx.x;
  int lane = tid & 31;
  int wave = tid >> 5;
  int wr = wave >> 1, wc = wave & 1;   // 4x2 wave grid; wave tile = 32x64
  int row0 = blockIdx.y * 128, col0 = blockIdx.x * 128;

  const v8f vz = {0, 0, 0, 0, 0, 0, 0, 0};
  v8f acc[2][4], acc2[2][4];
#pragma unroll
  for (int mi = 0; mi < 2; ++mi)
#pragma unroll
    for (int ni = 0; ni < 4; ++ni) { acc[mi][ni] = vz; if (MODE == 2) acc2[mi][ni] = vz; }

  int ar = tid >> 1, ac = (tid & 1) * 16;      // A tile 128x32: 16 elems / thread
  int br = tid >> 3, bc = (tid & 7) * 16;      // B tile 32x128: 16 elems / thread

  auto stage = [&](int buf, int k0) {
    int grow = row0 + ar;
    if (grow < Mrows) {
      uint32_t l = lds_off32(&As[buf][ar][ac]);
      const bf16_t* g = A + (size_t)grow * K + k0 + ac;
      async_g2l_b128(l, g);
      async_g2l_b128(l + 16, g + 8);
    } else {
#pragma unroll
      for (int i = 0; i < 16; ++i) As[buf][ar][ac + i] = (bf16_t)0.0f;
    }
    {
      uint32_t l = lds_off32(&Bs[buf][0][br][bc]);
      const bf16_t* g = B1 + (size_t)(k0 + br) * Ncols + col0 + bc;
      async_g2l_b128(l, g);
      async_g2l_b128(l + 16, g + 8);
    }
    if (MODE == 2) {
      uint32_t l = lds_off32(&Bs[buf][1][br][bc]);
      const bf16_t* g = B2p + (size_t)(k0 + br) * Ncols + col0 + bc;
      async_g2l_b128(l, g);
      async_g2l_b128(l + 16, g + 8);
    }
  };

  const int nk = K >> 5;
  stage(0, 0);
  for (int it = 0; it < nk; ++it) {
    int cur = it & 1;
    wait_asynccnt0();      // own async stores to LDS for buffer `cur` are done
    __syncthreads();       // everyone's are done; and everyone finished compute on cur^1
    if (it + 1 < nk) stage(cur ^ 1, (it + 1) << 5);

    v16bf a0 = frag_a_16x32(&As[cur][wr * 32][0], 40);
    v16bf a1 = frag_a_16x32(&As[cur][wr * 32 + 16][0], 40);
#pragma unroll
    for (int ni = 0; ni < 4; ++ni) {
      v16bf bfr = frag_b_32x16(&Bs[cur][0][0][wc * 64 + ni * 16], 136);
      acc[0][ni] = wmma_bf16(a0, bfr, acc[0][ni]);
      acc[1][ni] = wmma_bf16(a1, bfr, acc[1][ni]);
      if (MODE == 2) {
        v16bf b2 = frag_b_32x16(&Bs[cur][1][0][wc * 64 + ni * 16], 136);
        acc2[0][ni] = wmma_bf16(a0, b2, acc2[0][ni]);
        acc2[1][ni] = wmma_bf16(a1, b2, acc2[1][ni]);
      }
    }
  }

  int hi = lane >> 4, nl = lane & 15;
#pragma unroll
  for (int mi = 0; mi < 2; ++mi)
#pragma unroll
    for (int ni = 0; ni < 4; ++ni)
#pragma unroll
      for (int r = 0; r < 8; ++r) {
        int grow = row0 + wr * 32 + mi * 16 + r + 8 * hi;
        int gcol = col0 + wc * 64 + ni * 16 + nl;
        if (grow >= Mrows) continue;
        float v = acc[mi][ni][r];
        if (MODE == 0) {
          outf[(size_t)grow * Ncols + gcol] = v;
        } else if (MODE == 1) {
          int b = grow / N_;
          float g = mods[(size_t)b * SIXC_ + 2 * C_ + gcol];
          outf[(size_t)grow * Ncols + gcol] =
              resid[(size_t)grow * Ncols + gcol] + g * (v + bias[gcol]);
        } else if (MODE == 2) {
          float sil = v / (1.0f + __expf(-v));
          outb[(size_t)grow * Ncols + gcol] = (bf16_t)(sil * acc2[mi][ni][r]);
        } else {
          int b = grow / N_;
          float g = mods[(size_t)b * SIXC_ + 5 * C_ + gcol];
          outf[(size_t)grow * Ncols + gcol] = resid[(size_t)grow * Ncols + gcol] + g * v;
        }
      }
}

// ---------------- flash attention (bf16 WMMA, async double-buffered K/V) ----------------
// grid: (N_/64, B_*H_); block: 128 threads (4 waves, each wave owns 16 q rows)
__global__ __launch_bounds__(128, 1) void attn_kernel(const bf16_t* __restrict__ qb,
                                                      const bf16_t* __restrict__ kb,
                                                      const bf16_t* __restrict__ vb,
                                                      bf16_t* __restrict__ ob) {
  __shared__ __align__(16) bf16_t Ks[2][64][104];
  __shared__ __align__(16) bf16_t Vs[2][64][104];
  __shared__ __align__(16) bf16_t Ps[4][16][64];
  int tid = threadIdx.x;
  int lane = tid & 31, wave = tid >> 5;
  int hi = lane >> 4, nl = lane & 15;
  int bh = blockIdx.y;
  int b = bh / H_, h = bh % H_;
  int q0 = blockIdx.x * 64;
  const float scale = 0.1020620726f;   // 96^-0.5

  v16bf aq[3];
  const bf16_t* qbase = qb + ((size_t)bh * N_ + q0 + wave * 16) * HD_;
#pragma unroll
  for (int kk = 0; kk < 3; ++kk) aq[kk] = frag_a_16x32(qbase + kk * 32, HD_);

  float rowm[8], rowl[8];
  v8f o[6];
  const v8f vz = {0, 0, 0, 0, 0, 0, 0, 0};
#pragma unroll
  for (int r = 0; r < 8; ++r) { rowm[r] = -1e30f; rowl[r] = 0.0f; }
#pragma unroll
  for (int t = 0; t < 6; ++t) o[t] = vz;

  auto stageKV = [&](int buf, int kt) {
    int r = tid >> 1;
    int cb = (tid & 1) * 48;
    int key = kt * 64 + r;
    if (key < NK_) {
      const bf16_t* kg = kb + ((size_t)bh * NK_ + key) * HD_ + cb;
      const bf16_t* vg = vb + ((size_t)bh * NK_ + key) * HD_ + cb;
      uint32_t lk = lds_off32(&Ks[buf][r][cb]);
      uint32_t lv = lds_off32(&Vs[buf][r][cb]);
#pragma unroll
      for (int j = 0; j < 6; ++j) {
        async_g2l_b128(lk + j * 16, kg + j * 8);
        async_g2l_b128(lv + j * 16, vg + j * 8);
      }
    } else {
#pragma unroll
      for (int i = 0; i < 48; ++i) {
        Ks[buf][r][cb + i] = (bf16_t)0.0f;
        Vs[buf][r][cb + i] = (bf16_t)0.0f;
      }
    }
  };

  const int ntiles = (NK_ + 63) / 64;   // 34
  stageKV(0, 0);
  for (int kt = 0; kt < ntiles; ++kt) {
    int cur = kt & 1;
    wait_asynccnt0();
    __syncthreads();
    if (kt + 1 < ntiles) stageKV(cur ^ 1, kt + 1);

    // S = Q @ K^T  (16 q rows x 64 keys per wave)
    v8f s[4];
#pragma unroll
    for (int ni = 0; ni < 4; ++ni) s[ni] = vz;
#pragma unroll
    for (int kk = 0; kk < 3; ++kk)
#pragma unroll
      for (int ni = 0; ni < 4; ++ni) {
        v16bf bfr = frag_b_from_nt(&Ks[cur][ni * 16][kk * 32], 104);
        s[ni] = wmma_bf16(aq[kk], bfr, s[ni]);
      }

    // online softmax (row = r + 8*hi per lane-half; cols across 16 lanes)
#pragma unroll
    for (int r = 0; r < 8; ++r) {
      float sv[4];
      float mcur = -1e30f;
#pragma unroll
      for (int ni = 0; ni < 4; ++ni) {
        float v = s[ni][r] * scale;
        int kidx = kt * 64 + ni * 16 + nl;
        if (kidx >= NK_) v = -1e30f;
        sv[ni] = v;
        mcur = fmaxf(mcur, v);
      }
#pragma unroll
      for (int off = 8; off >= 1; off >>= 1) mcur = fmaxf(mcur, __shfl_xor(mcur, off));
      float mnew = fmaxf(rowm[r], mcur);
      float alpha = __expf(rowm[r] - mnew);
      rowm[r] = mnew;
      float rsum = 0.0f;
#pragma unroll
      for (int ni = 0; ni < 4; ++ni) {
        float p = __expf(sv[ni] - mnew);
        rsum += p;
        Ps[wave][r + 8 * hi][ni * 16 + nl] = (bf16_t)p;
      }
#pragma unroll
      for (int off = 8; off >= 1; off >>= 1) rsum += __shfl_xor(rsum, off);
      rowl[r] = rowl[r] * alpha + rsum;
#pragma unroll
      for (int t = 0; t < 6; ++t) o[t][r] *= alpha;
    }

    // O += P @ V   (per-wave P in LDS; same-wave DS ops are in order)
#pragma unroll
    for (int kk = 0; kk < 2; ++kk) {
      v16bf ap = frag_a_16x32(&Ps[wave][0][kk * 32], 64);
#pragma unroll
      for (int t = 0; t < 6; ++t) {
        v16bf bv = frag_b_32x16(&Vs[cur][kk * 32][t * 16], 104);
        o[t] = wmma_bf16(ap, bv, o[t]);
      }
    }
  }

  // write attn output in (B, N, H*HD) bf16 for the proj GEMM
#pragma unroll
  for (int t = 0; t < 6; ++t)
#pragma unroll
    for (int r = 0; r < 8; ++r) {
      int qrow = q0 + wave * 16 + r + 8 * hi;
      float val = o[t][r] / rowl[r];
      ob[((size_t)b * N_ + qrow) * C_ + h * HD_ + t * 16 + nl] = (bf16_t)val;
    }
}

// ---------------- host ----------------
static inline size_t align256(size_t x) { return (x + 255) & ~(size_t)255; }

extern "C" void kernel_launch(void* const* d_in, const int* in_sizes, int n_in,
                              void* d_out, int out_size, void* d_ws, size_t ws_size,
                              hipStream_t stream) {
  (void)in_sizes; (void)n_in; (void)out_size; (void)ws_size;
  const float* x        = (const float*)d_in[0];
  const float* y        = (const float*)d_in[1];
  const float* cvec     = (const float*)d_in[2];
  const float* rope_cos = (const float*)d_in[3];
  const float* rope_sin = (const float*)d_in[4];
  const float* w_qkv    = (const float*)d_in[5];
  const float* w_kv     = (const float*)d_in[6];
  const float* qn_w     = (const float*)d_in[7];
  const float* kn_w     = (const float*)d_in[8];
  const float* w_proj   = (const float*)d_in[9];
  const float* b_proj   = (const float*)d_in[10];
  const float* adaln_w  = (const float*)d_in[11];
  const float* adaln_b  = (const float*)d_in[12];
  const float* norm1_w  = (const float*)d_in[13];
  const float* norm2_w  = (const float*)d_in[14];
  const float* w1       = (const float*)d_in[15];
  const float* w3       = (const float*)d_in[16];
  const float* w2       = (const float*)d_in[17];
  float* out = (float*)d_out;

  char* p = (char*)d_ws;
  auto alloc = [&](size_t bytes) { char* r = p; p += align256(bytes); return r; };
  float*  mods     = (float*) alloc((size_t)B_ * SIXC_ * 4);
  bf16_t* h_bf     = (bf16_t*)alloc((size_t)BN_ * C_ * 2);        // reused later as h2
  bf16_t* wqkv_bf  = (bf16_t*)alloc((size_t)C_ * 3 * C_ * 2);
  bf16_t* wkv_bf   = (bf16_t*)alloc((size_t)C_ * 2 * C_ * 2);
  bf16_t* wproj_bf = (bf16_t*)alloc((size_t)C_ * C_ * 2);
  bf16_t* w1_bf    = (bf16_t*)alloc((size_t)C_ * FF_ * 2);
  bf16_t* w3_bf    = (bf16_t*)alloc((size_t)C_ * FF_ * 2);
  bf16_t* w2_bf    = (bf16_t*)alloc((size_t)FF_ * C_ * 2);
  bf16_t* y_bf     = (bf16_t*)alloc((size_t)B_ * M_ * C_ * 2);
  float*  qkv_f    = (float*) alloc((size_t)BN_ * 3 * C_ * 4);    // reused later as ff_bf
  float*  ykv_f    = (float*) alloc((size_t)B_ * M_ * 2 * C_ * 4);
  bf16_t* q_bf     = (bf16_t*)alloc((size_t)B_ * H_ * N_ * HD_ * 2);
  bf16_t* k_bf     = (bf16_t*)alloc((size_t)B_ * H_ * NK_ * HD_ * 2);
  bf16_t* v_bf     = (bf16_t*)alloc((size_t)B_ * H_ * NK_ * HD_ * 2);
  bf16_t* attn_bf  = (bf16_t*)alloc((size_t)BN_ * C_ * 2);
  float*  x2       = (float*) alloc((size_t)BN_ * C_ * 4);
  bf16_t* h2_bf    = h_bf;                 // alias: h dead after qkv GEMM
  bf16_t* ff_bf    = (bf16_t*)qkv_f;       // alias: qkv f32 dead after qkv_post

  auto cast = [&](const float* s, bf16_t* d, int n) {
    cast_bf16_kernel<<<(n + 255) / 256, 256, 0, stream>>>(s, d, n);
  };
  cast(w_qkv,  wqkv_bf,  C_ * 3 * C_);
  cast(w_kv,   wkv_bf,   C_ * 2 * C_);
  cast(w_proj, wproj_bf, C_ * C_);
  cast(w1,     w1_bf,    C_ * FF_);
  cast(w3,     w3_bf,    C_ * FF_);
  cast(w2,     w2_bf,    FF_ * C_);
  cast(y,      y_bf,     B_ * M_ * C_);

  adaln_kernel<<<(B_ * SIXC_ + 255) / 256, 256, 0, stream>>>(cvec, adaln_w, adaln_b, mods);

  modulate_kernel<<<BN_, 256, 0, stream>>>(x, norm1_w, mods, /*sh*/ 0, /*sc*/ C_, h_bf);

  // qkv = h @ w_qkv   (8192 x 3456 x 1152)
  gemm_bf16_kernel<0><<<dim3(3 * C_ / 128, BN_ / 128), 256, 0, stream>>>(
      h_bf, wqkv_bf, nullptr, BN_, 3 * C_, C_, qkv_f, nullptr, nullptr, nullptr, nullptr);

  // ykv = y @ w_kv    (400 x 2304 x 1152)
  gemm_bf16_kernel<0><<<dim3(2 * C_ / 128, (B_ * M_ + 127) / 128), 256, 0, stream>>>(
      y_bf, wkv_bf, nullptr, B_ * M_, 2 * C_, C_, ykv_f, nullptr, nullptr, nullptr, nullptr);

  qkv_post_kernel<<<BN_ * H_, 96, 0, stream>>>(qkv_f, qn_w, kn_w, rope_cos, rope_sin,
                                               q_bf, k_bf, v_bf);
  ykv_post_kernel<<<B_ * M_ * H_, 96, 0, stream>>>(ykv_f, kn_w, k_bf, v_bf);

  attn_kernel<<<dim3(N_ / 64, B_ * H_), 128, 0, stream>>>(q_bf, k_bf, v_bf, attn_bf);

  // x2 = x + g_msa * (attn @ w_proj + b_proj)
  gemm_bf16_kernel<1><<<dim3(C_ / 128, BN_ / 128), 256, 0, stream>>>(
      attn_bf, wproj_bf, nullptr, BN_, C_, C_, x2, nullptr, x, b_proj, mods);

  modulate_kernel<<<BN_, 256, 0, stream>>>(x2, norm2_w, mods, /*sh*/ 3 * C_, /*sc*/ 4 * C_, h2_bf);

  // ff = silu(h2 @ w1) * (h2 @ w3)
  gemm_bf16_kernel<2><<<dim3(FF_ / 128, BN_ / 128), 256, 0, stream>>>(
      h2_bf, w1_bf, w3_bf, BN_, FF_, C_, nullptr, ff_bf, nullptr, nullptr, nullptr);

  // out = x2 + g_mlp * (ff @ w2)
  gemm_bf16_kernel<3><<<dim3(C_ / 128, BN_ / 128), 256, 0, stream>>>(
      ff_bf, w2_bf, nullptr, BN_, C_, FF_, out, nullptr, x2, nullptr, mods);
}
